// LatentHamiltonianPredictor_14207751815608
// MI455X (gfx1250) — matrix operand order repaired and checked by reading, not compiled, not run on hardware
//
#include <hip/hip_runtime.h>
#include <hip/hip_bf16.h>
#include <math.h>
#include <stdint.h>

#define HALF_DIM 128
#define HIDDEN   2048
#define EMB_DIM  32

typedef __attribute__((ext_vector_type(16))) __bf16 v16bf;
typedef __attribute__((ext_vector_type(8)))  float  v8f;
typedef __attribute__((ext_vector_type(4)))  int    v4i;

union FragAB { v16bf v; uint4 q[2]; };
union FragC  { v8f   v; float f[8]; };

__device__ __forceinline__ unsigned short f32_to_bf16(float f) {
  unsigned int x = __float_as_uint(f);
  x += 0x7fffu + ((x >> 16) & 1u);          // round-to-nearest-even
  return (unsigned short)(x >> 16);
}
__device__ __forceinline__ float bf16_to_f32(unsigned short h) {
  return __uint_as_float(((unsigned int)h) << 16);
}

// A-fragment (16x32 bf16, M x K): lane m = lane&15, half = lane>>4.
// VGPRs 0..3 hold K = kb + 8*half + {0..7}; VGPRs 4..7 hold K = kb+16+8*half+{0..7}.
__device__ __forceinline__ FragAB load_frag_a(const unsigned short* base, int row,
                                              int ld, int kb, int half) {
  FragAB f;
  const unsigned short* p = base + row * ld + kb + 8 * half;
  f.q[0] = *reinterpret_cast<const uint4*>(p);
  f.q[1] = *reinterpret_cast<const uint4*>(p + 16);
  return f;
}

// B-fragment (32x16 bf16, K x N): lane n = lane&15, half = lane>>4.
// VGPRs 0..3 hold K = kb + 8*half + {0..7}; VGPRs 4..7 hold K = kb+16+8*half+{0..7}
// (B column n == source row n over K).
__device__ __forceinline__ FragAB load_frag_b(const unsigned short* base, int row,
                                              int ld, int kb, int half) {
  FragAB f;
  const unsigned short* p = base + row * ld + kb + 8 * half;
  f.q[0] = *reinterpret_cast<const uint4*>(p);
  f.q[1] = *reinterpret_cast<const uint4*>(p + 16);
  return f;
}

#define WMMA_BF16(A, B, C) \
  __builtin_amdgcn_wmma_f32_16x16x32_bf16(false, (A), false, (B), (short)0, (C), false, false)

// ---------------------------------------------------------------------------
// Async global->LDS path (gfx1250 ASYNCcnt). Guarded: falls back to the
// synchronous staging loop if the toolchain lacks the builtins.
// Signature (from hipcc diagnostic): (v4i AS1* global_src, v4i AS3* lds_dst,
// imm offset, imm cpol).
// ---------------------------------------------------------------------------
#if __has_builtin(__builtin_amdgcn_global_load_async_to_lds_b128)
#define LH_ASYNC 1
#else
#define LH_ASYNC 0
#endif

#if LH_ASYNC
__device__ __forceinline__ void lh_async_b128(const unsigned short* g,
                                              unsigned short* l) {
  // Global generic pointer bit-pattern == AS1 pointer; generic LDS pointer's
  // low 32 bits == wave-relative LDS byte offset (AS3 representation).
  __builtin_amdgcn_global_load_async_to_lds_b128(
      (__attribute__((address_space(1))) v4i*)(uintptr_t)g,
      (__attribute__((address_space(3))) v4i*)(unsigned int)(uintptr_t)l,
      0, 0);
}
__device__ __forceinline__ void lh_wait_async() {
#if __has_builtin(__builtin_amdgcn_s_wait_asynccnt)
  __builtin_amdgcn_s_wait_asynccnt(0);
#else
  asm volatile("s_wait_asynccnt 0x0" ::: "memory");
#endif
}
#define LH_WBUF 2
#else
#define LH_WBUF 1
#endif

// ---------------------------------------------------------------------------
// Split fp32 weight (HIDDEN x HALF_DIM) into bf16 hi/lo, row-major and
// transposed (HALF_DIM x HIDDEN) copies for the two GEMM orientations.
// ---------------------------------------------------------------------------
__global__ void lh_split_weight(const float* __restrict__ W,
                                unsigned short* __restrict__ Whi,
                                unsigned short* __restrict__ Wlo,
                                unsigned short* __restrict__ WThi,
                                unsigned short* __restrict__ WTlo) {
  int i = blockIdx.x * 256 + threadIdx.x;
  if (i >= HIDDEN * HALF_DIM) return;
  int rr = i / HALF_DIM, cc = i % HALF_DIM;
  float v = W[i];
  unsigned short h = f32_to_bf16(v);
  unsigned short l = f32_to_bf16(v - bf16_to_f32(h));
  Whi[i] = h;  Wlo[i] = l;
  WThi[cc * HIDDEN + rr] = h;
  WTlo[cc * HIDDEN + rr] = l;
}

// ---------------------------------------------------------------------------
// G_u[n,d] = emb_table[action[n]] . GW[d,:] + Gb[d]   (tiny, K=32)
// ---------------------------------------------------------------------------
__global__ void lh_gu_kernel(const int* __restrict__ action,
                             const float* __restrict__ emb,
                             const float* __restrict__ GW,
                             const float* __restrict__ Gb,
                             float* __restrict__ Gu, int Ntot) {
  int i = blockIdx.x * 256 + threadIdx.x;
  if (i >= Ntot * HALF_DIM) return;
  int n = i >> 7, d = i & 127;
  const float* e = emb + action[n] * EMB_DIM;
  const float* g = GW + d * EMB_DIM;
  float s = Gb[d];
#pragma unroll
  for (int k = 0; k < EMB_DIM; ++k) s += e[k] * g[k];
  Gu[i] = s;
}

// ---------------------------------------------------------------------------
// Fused MLP gradient: out[n,:] = (sigmoid(x[n,:]@W1^T + b1) * w2) @ W1
// 1 workgroup = 8 waves = 128 rows; HIDDEN walked in chunks of 32 with
// double-buffered async weight staging (depth-1 pipeline).
// ---------------------------------------------------------------------------
__global__ __launch_bounds__(256) void lh_mlp_grad(
    const float* __restrict__ x, int xstride, int xoff,
    const unsigned short* __restrict__ Whi, const unsigned short* __restrict__ Wlo,
    const unsigned short* __restrict__ WThi, const unsigned short* __restrict__ WTlo,
    const float* __restrict__ b1, const float* __restrict__ w2,
    float* __restrict__ out, int Ntot) {
  __shared__ __align__(16) unsigned short xs_hi[128 * 128];           // 32 KB
  __shared__ __align__(16) unsigned short xs_lo[128 * 128];           // 32 KB
  __shared__ __align__(16) unsigned short wr_hi[LH_WBUF][32 * 128];   // 8 KB/buf
  __shared__ __align__(16) unsigned short wr_lo[LH_WBUF][32 * 128];
  __shared__ __align__(16) unsigned short wt_hi[LH_WBUF][128 * 32];
  __shared__ __align__(16) unsigned short wt_lo[LH_WBUF][128 * 32];
  __shared__ __align__(16) unsigned short sc_hi[8 * 16 * 32];         // 8 KB
  __shared__ __align__(16) unsigned short sc_lo[8 * 16 * 32];

  const int tid  = threadIdx.x;
  const int wave = tid >> 5;
  const int lane = tid & 31;
  const int half = lane >> 4;
  const int r16  = lane & 15;
  const int row0 = blockIdx.x * 128;
  const int m0   = wave * 16;

  // Stage x tile (128 rows x 128 cols), split into bf16 hi/lo.
  for (int i = tid; i < 128 * 128; i += 256) {
    int rr = i >> 7, cc = i & 127;
    int g = row0 + rr;
    float v = (g < Ntot) ? x[(size_t)g * xstride + xoff + cc] : 0.0f;
    unsigned short h = f32_to_bf16(v);
    xs_hi[i] = h;
    xs_lo[i] = f32_to_bf16(v - bf16_to_f32(h));
  }

#if LH_ASYNC
  // Prologue: kick off chunk 0 weight staging (8x b128 per thread).
#pragma unroll
  for (int i = 0; i < 2; ++i) {
    int idx  = tid + 256 * i;        // 0..511 uint4 tiles
    int rowd = idx >> 2;
    int qi   = idx & 3;
    lh_async_b128(Whi  + (size_t)idx * 8,                  &wr_hi[0][idx * 8]);
    lh_async_b128(Wlo  + (size_t)idx * 8,                  &wr_lo[0][idx * 8]);
    lh_async_b128(WThi + (size_t)rowd * HIDDEN + qi * 8,   &wt_hi[0][idx * 8]);
    lh_async_b128(WTlo + (size_t)rowd * HIDDEN + qi * 8,   &wt_lo[0][idx * 8]);
  }
#endif

  FragC gacc[8];
#pragma unroll
  for (int d = 0; d < 8; ++d)
#pragma unroll
    for (int e = 0; e < 8; ++e) gacc[d].f[e] = 0.0f;

  unsigned short* mys_hi = sc_hi + wave * (16 * 32);
  unsigned short* mys_lo = sc_lo + wave * (16 * 32);

  for (int cidx = 0; cidx < HIDDEN / 32; ++cidx) {
    const int jc  = cidx * 32;
#if LH_ASYNC
    const int cur = cidx & 1;
    lh_wait_async();     // my staging ops for chunk `cidx` complete
    __syncthreads();     // everyone's complete; spare buffer now free
    if (jc + 32 < HIDDEN) {
      const int nxt = cur ^ 1;
      const size_t jn = (size_t)(jc + 32);
#pragma unroll
      for (int i = 0; i < 2; ++i) {
        int idx  = tid + 256 * i;
        int rowd = idx >> 2;
        int qi   = idx & 3;
        lh_async_b128(Whi  + jn * 128 + idx * 8,                  &wr_hi[nxt][idx * 8]);
        lh_async_b128(Wlo  + jn * 128 + idx * 8,                  &wr_lo[nxt][idx * 8]);
        lh_async_b128(WThi + (size_t)rowd * HIDDEN + jn + qi * 8, &wt_hi[nxt][idx * 8]);
        lh_async_b128(WTlo + (size_t)rowd * HIDDEN + jn + qi * 8, &wt_lo[nxt][idx * 8]);
      }
    }
#else
    const int cur = 0;
    __syncthreads();
    {
      const uint4* shi = reinterpret_cast<const uint4*>(Whi + (size_t)jc * 128);
      const uint4* slo = reinterpret_cast<const uint4*>(Wlo + (size_t)jc * 128);
      uint4* dhi = reinterpret_cast<uint4*>(wr_hi[0]);
      uint4* dlo = reinterpret_cast<uint4*>(wr_lo[0]);
#pragma unroll
      for (int i = 0; i < 2; ++i) {
        dhi[tid + 256 * i] = shi[tid + 256 * i];
        dlo[tid + 256 * i] = slo[tid + 256 * i];
      }
#pragma unroll
      for (int i = 0; i < 2; ++i) {
        int idx  = tid + 256 * i;
        int rowd = idx >> 2;
        int qi   = idx & 3;
        reinterpret_cast<uint4*>(wt_hi[0])[idx] =
            *reinterpret_cast<const uint4*>(WThi + (size_t)rowd * HIDDEN + jc + qi * 8);
        reinterpret_cast<uint4*>(wt_lo[0])[idx] =
            *reinterpret_cast<const uint4*>(WTlo + (size_t)rowd * HIDDEN + jc + qi * 8);
      }
      if (jc + 32 < HIDDEN) {
        __builtin_prefetch(Whi + (size_t)(jc + 32) * 128 + tid * 16, 0, 1);
        __builtin_prefetch(Wlo + (size_t)(jc + 32) * 128 + tid * 16, 0, 1);
      }
    }
    __syncthreads();
#endif

    // --- GEMM1: h(16x32) = x(16x128) @ W1_chunk^T, bf16x2 split (3 terms) ---
    FragC hacc[2];
#pragma unroll
    for (int nt = 0; nt < 2; ++nt)
#pragma unroll
      for (int e = 0; e < 8; ++e) hacc[nt].f[e] = 0.0f;

#pragma unroll
    for (int ks = 0; ks < 4; ++ks) {
      FragAB axh = load_frag_a(xs_hi, m0 + r16, 128, ks * 32, half);
      FragAB axl = load_frag_a(xs_lo, m0 + r16, 128, ks * 32, half);
#pragma unroll
      for (int nt = 0; nt < 2; ++nt) {
        FragAB bwh = load_frag_b(wr_hi[cur], nt * 16 + r16, 128, ks * 32, half);
        FragAB bwl = load_frag_b(wr_lo[cur], nt * 16 + r16, 128, ks * 32, half);
        hacc[nt].v = WMMA_BF16(axh.v, bwh.v, hacc[nt].v);
        hacc[nt].v = WMMA_BF16(axh.v, bwl.v, hacc[nt].v);
        hacc[nt].v = WMMA_BF16(axl.v, bwh.v, hacc[nt].v);
      }
    }

    // --- s = sigmoid(h + b1) * w2 ; split to bf16 hi/lo in per-wave scratch ---
#pragma unroll
    for (int nt = 0; nt < 2; ++nt) {
      int j = jc + nt * 16 + r16;
      float b1v = b1[j];
      float w2v = w2[j];
#pragma unroll
      for (int e = 0; e < 8; ++e) {
        int m = e + 8 * half;          // C-layout: VGPR e holds row m
        float hv = hacc[nt].f[e] + b1v;
        float sv = w2v / (1.0f + __expf(-hv));
        unsigned short sh = f32_to_bf16(sv);
        mys_hi[m * 32 + nt * 16 + r16] = sh;
        mys_lo[m * 32 + nt * 16 + r16] = f32_to_bf16(sv - bf16_to_f32(sh));
      }
    }
    // per-wave scratch: DS ops are in-order within a wave, no barrier needed

    // --- GEMM2: grad(16x128) += s(16x32) @ W1_chunk(32x128) ---
    FragAB ash = load_frag_a(mys_hi, r16, 32, 0, half);
    FragAB asl = load_frag_a(mys_lo, r16, 32, 0, half);
#pragma unroll
    for (int dt = 0; dt < 8; ++dt) {
      FragAB bh = load_frag_b(wt_hi[cur], dt * 16 + r16, 32, 0, half);
      FragAB bl = load_frag_b(wt_lo[cur], dt * 16 + r16, 32, 0, half);
      gacc[dt].v = WMMA_BF16(ash.v, bh.v, gacc[dt].v);
      gacc[dt].v = WMMA_BF16(ash.v, bl.v, gacc[dt].v);
      gacc[dt].v = WMMA_BF16(asl.v, bh.v, gacc[dt].v);
    }
  }

  // --- Write grad tile (fp32, N x 128 contiguous) ---
#pragma unroll
  for (int dt = 0; dt < 8; ++dt)
#pragma unroll
    for (int e = 0; e < 8; ++e) {
      int m = e + 8 * half;
      int g = row0 + m0 + m;
      if (g < Ntot) out[(size_t)g * 128 + dt * 16 + r16] = gacc[dt].f[e];
    }
}

// ---------------------------------------------------------------------------
// Leapfrog elementwise steps (DT = 1.0)
// ---------------------------------------------------------------------------
__global__ void lh_combine1(const float* __restrict__ context,
                            const float* __restrict__ dHq,
                            const float* __restrict__ dHp,
                            const float* __restrict__ Gu,
                            const float* __restrict__ log_damping,
                            float* __restrict__ p_half, int Ntot) {
  int i = blockIdx.x * 256 + threadIdx.x;
  if (i >= Ntot * HALF_DIM) return;
  int n = i >> 7, d = i & 127;
  float damping = __expf(log_damping[0]);
  float p = context[(size_t)n * 256 + 128 + d];
  p_half[i] = p + 0.5f * (-dHq[i] - damping * dHp[i] + Gu[i]);
}

__global__ void lh_combine2(const float* __restrict__ context,
                            const float* __restrict__ dHp_half,
                            float* __restrict__ q_next,
                            float* __restrict__ out, int Ntot) {
  int i = blockIdx.x * 256 + threadIdx.x;
  if (i >= Ntot * HALF_DIM) return;
  int n = i >> 7, d = i & 127;
  float qn = context[(size_t)n * 256 + d] + dHp_half[i];
  q_next[i] = qn;
  out[(size_t)n * 256 + d] = qn;
}

__global__ void lh_combine3(const float* __restrict__ p_half,
                            const float* __restrict__ dHq_next,
                            const float* __restrict__ dHp_half,
                            const float* __restrict__ Gu,
                            const float* __restrict__ log_damping,
                            float* __restrict__ out, int Ntot) {
  int i = blockIdx.x * 256 + threadIdx.x;
  if (i >= Ntot * HALF_DIM) return;
  int n = i >> 7, d = i & 127;
  float damping = __expf(log_damping[0]);
  out[(size_t)n * 256 + 128 + d] =
      p_half[i] + 0.5f * (-dHq_next[i] - damping * dHp_half[i] + Gu[i]);
}

// ---------------------------------------------------------------------------
extern "C" void kernel_launch(void* const* d_in, const int* in_sizes, int n_in,
                              void* d_out, int out_size, void* d_ws, size_t ws_size,
                              hipStream_t stream) {
  const float* context     = (const float*)d_in[0];
  const int*   action      = (const int*)  d_in[1];
  const float* emb_table   = (const float*)d_in[2];
  const float* TW1         = (const float*)d_in[3];
  const float* Tb1         = (const float*)d_in[4];
  const float* TW2         = (const float*)d_in[5];
  const float* VW1         = (const float*)d_in[7];
  const float* Vb1         = (const float*)d_in[8];
  const float* VW2         = (const float*)d_in[9];
  const float* GW          = (const float*)d_in[11];
  const float* Gb          = (const float*)d_in[12];
  const float* log_damping = (const float*)d_in[13];
  float* out = (float*)d_out;

  const int Ntot = in_sizes[0] / (2 * HALF_DIM);

  // Workspace carving (256B aligned)
  char* ws = (char*)d_ws;
  size_t off = 0;
  auto carve = [&](size_t bytes) {
    void* p = ws + off;
    off += (bytes + 255) & ~(size_t)255;
    return p;
  };
  const size_t wsz = (size_t)HIDDEN * HALF_DIM * sizeof(unsigned short);
  unsigned short* vhi  = (unsigned short*)carve(wsz);
  unsigned short* vlo  = (unsigned short*)carve(wsz);
  unsigned short* vthi = (unsigned short*)carve(wsz);
  unsigned short* vtlo = (unsigned short*)carve(wsz);
  unsigned short* thi  = (unsigned short*)carve(wsz);
  unsigned short* tlo  = (unsigned short*)carve(wsz);
  unsigned short* tthi = (unsigned short*)carve(wsz);
  unsigned short* ttlo = (unsigned short*)carve(wsz);
  const size_t fsz = (size_t)Ntot * HALF_DIM * sizeof(float);
  float* Gu   = (float*)carve(fsz);
  float* bufA = (float*)carve(fsz);   // dHq, later dHq_next
  float* bufB = (float*)carve(fsz);   // dHp
  float* bufC = (float*)carve(fsz);   // p_half
  float* bufD = (float*)carve(fsz);   // dHp_half
  float* bufE = (float*)carve(fsz);   // q_next

  const int ew_grid   = (Ntot * HALF_DIM + 255) / 256;
  const int grad_grid = (Ntot + 127) / 128;
  const int sw_grid   = (HIDDEN * HALF_DIM + 255) / 256;

  // 1) pre-split weights to bf16 hi/lo (+ transposed copies)
  lh_split_weight<<<sw_grid, 256, 0, stream>>>(VW1, vhi, vlo, vthi, vtlo);
  lh_split_weight<<<sw_grid, 256, 0, stream>>>(TW1, thi, tlo, tthi, ttlo);

  // 2) G_u
  lh_gu_kernel<<<ew_grid, 256, 0, stream>>>(action, emb_table, GW, Gb, Gu, Ntot);

  // 3) dH_dq = grad_V(q); q = context[..., :128]
  lh_mlp_grad<<<grad_grid, 256, 0, stream>>>(context, 256, 0,
      vhi, vlo, vthi, vtlo, Vb1, VW2, bufA, Ntot);
  // 4) dH_dp = grad_T(p); p = context[..., 128:]
  lh_mlp_grad<<<grad_grid, 256, 0, stream>>>(context, 256, 128,
      thi, tlo, tthi, ttlo, Tb1, TW2, bufB, Ntot);

  // 5) p_half
  lh_combine1<<<ew_grid, 256, 0, stream>>>(context, bufA, bufB, Gu, log_damping,
                                           bufC, Ntot);

  // 6) dH_dp_half = grad_T(p_half)
  lh_mlp_grad<<<grad_grid, 256, 0, stream>>>(bufC, 128, 0,
      thi, tlo, tthi, ttlo, Tb1, TW2, bufD, Ntot);

  // 7) q_next (also writes output q half)
  lh_combine2<<<ew_grid, 256, 0, stream>>>(context, bufD, bufE, out, Ntot);

  // 8) dH_dq_next = grad_V(q_next)
  lh_mlp_grad<<<grad_grid, 256, 0, stream>>>(bufE, 128, 0,
      vhi, vlo, vthi, vtlo, Vb1, VW2, bufA, Ntot);

  // 9) p_next (output p half)
  lh_combine3<<<ew_grid, 256, 0, stream>>>(bufC, bufA, bufD, Gu, log_damping,
                                           out, Ntot);
}
